// GNN_2911987826770
// MI455X (gfx1250) — matrix operand order repaired and checked by reading, not compiled, not run on hardware
//
#include <hip/hip_runtime.h>
#include <hip/hip_bf16.h>

#define NND 20000     // nodes
#define NE  320000    // edges
#define DM  512       // feature dim
#define NL  6         // layers
#define NG  64        // graphs
#define NC  10        // classes
#define EPSV 1e-5f

typedef __bf16 bf16_t;
typedef __attribute__((ext_vector_type(16))) __bf16 v16bf;
typedef __attribute__((ext_vector_type(8)))  __bf16 v8bf;
typedef __attribute__((ext_vector_type(4)))  __bf16 v4bf;
typedef __attribute__((ext_vector_type(8)))  float  v8f;
typedef __attribute__((ext_vector_type(4)))  unsigned int v4u;
typedef __attribute__((ext_vector_type(4)))  int v4i;
typedef __attribute__((ext_vector_type(8)))  int v8i;

#define SA_STRIDE 520   // padded bf16 row stride in LDS (bank-conflict free)

#if __has_builtin(__builtin_amdgcn_tensor_load_to_lds)
#define HAVE_TDM 1
#else
#define HAVE_TDM 0
#endif

// ---------------------------------------------------------------- utilities
__global__ __launch_bounds__(256) void zero_f4(float4* __restrict__ p, int n4) {
    int i = blockIdx.x * 256 + threadIdx.x;
    if (i < n4) p[i] = make_float4(0.f, 0.f, 0.f, 0.f);
}

// ------------------------------------------------- pack weights to bf16 WMMA
// Packed layout per 512x512 matrix: idx = (((tk*32 + tn)*32 + lane)*16 + j)
//   half = lane>>4, c = lane&15 ; element j holds W[tk*32 + (j<8 ? half*8 + j : 16 + half*8 + j-8)][tn*16 + c]
__global__ __launch_bounds__(256)
void pack_all(const float* __restrict__ Wrel, const float* __restrict__ Wroot,
              bf16_t* __restrict__ P) {
    int idx = blockIdx.x * 256 + threadIdx.x;          // 12 * 2^18 total
    if (idx >= 12 * (1 << 18)) return;
    int within = idx & ((1 << 18) - 1);
    int m    = idx >> 18;
    int j    = within & 15;
    int lane = (within >> 4) & 31;
    int tn   = (within >> 9) & 31;
    int tk   = (within >> 14) & 15;
    int half = lane >> 4, c = lane & 15;
    int k   = tk * 32 + (j < 8 ? half * 8 + j : 16 + half * 8 + (j - 8));
    int col = tn * 16 + c;
    const float* W = (m < 6) ? (Wrel + (size_t)m * DM * DM)
                             : (Wroot + (size_t)(m - 6) * DM * DM);
    P[idx] = (bf16_t)W[(size_t)k * DM + col];
}

// --------------------------------------------------------- edge scatter-add
__global__ __launch_bounds__(256)
void edge_scatter(const float* __restrict__ x, const int* __restrict__ src,
                  const int* __restrict__ dst, float* __restrict__ agg) {
    int gid = blockIdx.x * 256 + threadIdx.x;          // NE * 128 threads
    if (gid >= NE * 128) return;
    int e = gid >> 7, c = gid & 127;
    int s = src[e], d = dst[e];
    float4 v = ((const float4*)(x + (size_t)s * DM))[c];
    float* ap = agg + (size_t)d * DM + c * 4;
    unsafeAtomicAdd(ap + 0, v.x);
    unsafeAtomicAdd(ap + 1, v.y);
    unsafeAtomicAdd(ap + 2, v.z);
    unsafeAtomicAdd(ap + 3, v.w);
}

// ------------------------------------------------ TDM 8KB contiguous -> LDS
#if HAVE_TDM
__device__ __forceinline__ void tdm_copy_8k(const bf16_t* gsrc, bf16_t* ldst) {
    unsigned long long ga = (unsigned long long)(uintptr_t)gsrc;
    unsigned lo = (unsigned)(uintptr_t)ldst;           // low 32b = LDS byte offset
    v4u g0;
    g0[0] = 1u;                                        // count=1, user mode
    g0[1] = lo;                                        // lds_addr
    g0[2] = (unsigned)(ga & 0xFFFFFFFFu);              // global_addr[31:0]
    g0[3] = (unsigned)((ga >> 32) & 0x1FFFFFFu)        // global_addr[56:32]
          | (2u << 30);                                // type = 2 (image)
    v8i g1;
    g1[0] = 3 << 16;                                   // data_size = 8 bytes
    g1[1] = (int)(1024u << 16);                        // tensor_dim0 = 1024 (lo16)
    g1[2] = 1 << 16;                                   // dim0 hi = 0 | tensor_dim1 = 1
    g1[3] = (int)(1024u << 16);                        // tile_dim0 = 1024
    g1[4] = 1;                                         // tile_dim1 = 1, tile_dim2 = 0
    g1[5] = 1024;                                      // tensor_dim0_stride = 1024
    g1[6] = 0;
    g1[7] = 0;
    v4i z4 = {0, 0, 0, 0};
    v8i z8 = {0, 0, 0, 0, 0, 0, 0, 0};
    // 6-arg form (clang-23 / therock headers): (g0, g1, g2, g3, g4, cpol)
    __builtin_amdgcn_tensor_load_to_lds(g0, g1, z4, z4, z8, 0);
}
#endif

// stage 8KB of packed B (one K-step, 8 tn tiles) per matrix into LDS
__device__ __forceinline__ void stage_B(const bf16_t* __restrict__ Prel,
                                        const bf16_t* __restrict__ Proot,
                                        int tk, int tn0,
                                        bf16_t* dRel, bf16_t* dRoot) {
#if HAVE_TDM
    if (threadIdx.x < 32) {                            // wave 0 drives the TDM
        tdm_copy_8k(Prel  + (size_t)(tk * 32 + tn0) * 512, dRel);
        tdm_copy_8k(Proot + (size_t)(tk * 32 + tn0) * 512, dRoot);
    }
#else
    const uint4* s0 = (const uint4*)(Prel  + (size_t)(tk * 32 + tn0) * 512);
    const uint4* s1 = (const uint4*)(Proot + (size_t)(tk * 32 + tn0) * 512);
    uint4* d0 = (uint4*)dRel;
    uint4* d1 = (uint4*)dRoot;
    for (int i = threadIdx.x; i < 512; i += 256) { d0[i] = s0[i]; d1[i] = s1[i]; }
#endif
}

// ---------------------------------------------------- WMMA fused dual GEMM
// h = agg @ Wrel + x @ Wroot + b_rel ; also accumulate per-column sum / sumsq
__global__ __launch_bounds__(256)
void gemm_wmma(const float* __restrict__ agg, const float* __restrict__ x,
               const bf16_t* __restrict__ Prel, const bf16_t* __restrict__ Proot,
               const float* __restrict__ brel, float* __restrict__ h,
               float* __restrict__ colsum, float* __restrict__ colsq) {
    __shared__ alignas(32) bf16_t sA[2][16 * SA_STRIDE];   // [mat][row*520 + k]
    __shared__ alignas(32) bf16_t sB[2][2][4096];          // [buf][mat][tile data]

    int wave = threadIdx.x >> 5;
    int lane = threadIdx.x & 31;
    int tm   = blockIdx.x >> 2;                 // 1250 row tiles
    int tn0  = (blockIdx.x & 3) * 8;            // 8 col tiles per block
    int tn   = tn0 + wave;
    int half = lane >> 4;
    int c    = lane & 15;

    // ---- cooperative A stage: 16 rows of agg and x -> bf16 in LDS ----
    const float4* ag4 = (const float4*)(agg + (size_t)tm * 16 * DM);
    const float4* xx4 = (const float4*)(x   + (size_t)tm * 16 * DM);
    for (int i = threadIdx.x; i < 2048; i += 256) {        // 16*512/4 float4s
        int r  = i >> 7;                                   // row
        int kq = i & 127;                                  // float4 within row
        if (i + 256 < 2048) __builtin_prefetch(&ag4[i + 256], 0, 0);
        float4 a = ag4[i], b = xx4[i];
        v4bf va; va[0]=(bf16_t)a.x; va[1]=(bf16_t)a.y; va[2]=(bf16_t)a.z; va[3]=(bf16_t)a.w;
        v4bf vb; vb[0]=(bf16_t)b.x; vb[1]=(bf16_t)b.y; vb[2]=(bf16_t)b.z; vb[3]=(bf16_t)b.w;
        *(v4bf*)&sA[0][r * SA_STRIDE + kq * 4] = va;
        *(v4bf*)&sA[1][r * SA_STRIDE + kq * 4] = vb;
    }

    stage_B(Prel, Proot, 0, tn0, sB[0][0], sB[0][1]);

    v8f acc = {};
    for (int tk = 0; tk < 16; ++tk) {
        int buf = tk & 1;
        if (tk < 15) {
            stage_B(Prel, Proot, tk + 1, tn0, sB[buf ^ 1][0], sB[buf ^ 1][1]);
            __builtin_amdgcn_s_wait_tensorcnt(2);          // step-tk pair landed
        } else {
            __builtin_amdgcn_s_wait_tensorcnt(0);
        }
        __syncthreads();                                   // A + B(tk) visible

        int kk = tk * 32;
        // A fragments: lane holds row c, k-offsets {half*8..+7} and {half*8+16..+23}
        const v8bf* aA = (const v8bf*)&sA[0][c * SA_STRIDE + kk + half * 8];
        const v8bf* aX = (const v8bf*)&sA[1][c * SA_STRIDE + kk + half * 8];
        v8bf alo = aA[0], ahi = aA[2];
        v8bf xlo = aX[0], xhi = aX[2];
        v16bf A = __builtin_shufflevector(alo, ahi, 0,1,2,3,4,5,6,7,8,9,10,11,12,13,14,15);
        v16bf X = __builtin_shufflevector(xlo, xhi, 0,1,2,3,4,5,6,7,8,9,10,11,12,13,14,15);
        // B fragments: each lane's v16bf is 32B contiguous in the staged tile
        v16bf Br = ((const v16bf*)sB[buf][0])[wave * 32 + lane];
        v16bf Bo = ((const v16bf*)sB[buf][1])[wave * 32 + lane];

        acc = __builtin_amdgcn_wmma_f32_16x16x32_bf16(false, A, false, Br,
                                                      (short)0, acc, false, false);
        acc = __builtin_amdgcn_wmma_f32_16x16x32_bf16(false, X, false, Bo,
                                                      (short)0, acc, false, false);
        __syncthreads();                                   // done reading buf
    }

    // C/D layout: lanes 0-15 hold M = i, lanes 16-31 hold M = 8+i (VGPR i); N = lane&15
    int coln = tn * 16 + c;
    float bias = brel[coln];
    int rbase = tm * 16 + half * 8;
    float s = 0.f, s2 = 0.f;
    #pragma unroll
    for (int i = 0; i < 8; ++i) {
        float v = acc[i] + bias;
        h[(size_t)(rbase + i) * DM + coln] = v;
        s += v; s2 += v * v;
    }
    unsafeAtomicAdd(&colsum[coln], s);
    unsafeAtomicAdd(&colsq[coln], s2);
}

// ----------------------------------------------------------- BN finalize/apply
__global__ void bn_finalize(const float* __restrict__ colsum, const float* __restrict__ colsq,
                            const float* __restrict__ gamma, const float* __restrict__ beta,
                            float* __restrict__ scale, float* __restrict__ shift) {
    int d = blockIdx.x * blockDim.x + threadIdx.x;
    if (d >= DM) return;
    float mu  = colsum[d] * (1.0f / NND);
    float var = colsq[d] * (1.0f / NND) - mu * mu;
    float sc  = gamma[d] * rsqrtf(var + EPSV);
    scale[d] = sc;
    shift[d] = beta[d] - mu * sc;
}

__global__ __launch_bounds__(256)
void bn_apply(const float* __restrict__ h, const float* __restrict__ scale,
              const float* __restrict__ shift, float* __restrict__ xout) {
    int i = blockIdx.x * 256 + threadIdx.x;            // NND*DM/4
    if (i >= NND * DM / 4) return;
    int d0 = (i * 4) & (DM - 1);
    float4 v = ((const float4*)h)[i];
    float4 r;
    r.x = fmaxf(v.x * scale[d0 + 0] + shift[d0 + 0], 0.f);
    r.y = fmaxf(v.y * scale[d0 + 1] + shift[d0 + 1], 0.f);
    r.z = fmaxf(v.z * scale[d0 + 2] + shift[d0 + 2], 0.f);
    r.w = fmaxf(v.w * scale[d0 + 3] + shift[d0 + 3], 0.f);
    ((float4*)xout)[i] = r;
}

// ----------------------------------------------------------------- pooling
__global__ __launch_bounds__(256)
void pool_accum(const float* __restrict__ x, const int* __restrict__ batch,
                float* __restrict__ sums, float* __restrict__ counts) {
    int gid = blockIdx.x * 256 + threadIdx.x;          // NND * 128
    if (gid >= NND * 128) return;
    int n = gid >> 7, cidx = gid & 127;
    int g = batch[n];
    float4 v = ((const float4*)(x + (size_t)n * DM))[cidx];
    float* p = sums + (size_t)g * DM + cidx * 4;
    unsafeAtomicAdd(p + 0, v.x);
    unsafeAtomicAdd(p + 1, v.y);
    unsafeAtomicAdd(p + 2, v.z);
    unsafeAtomicAdd(p + 3, v.w);
    if (cidx == 0) unsafeAtomicAdd(&counts[g], 1.0f);
}

__global__ void head(const float* __restrict__ sums, const float* __restrict__ counts,
                     const float* __restrict__ lw, const float* __restrict__ lb,
                     float* __restrict__ out) {
    int i = blockIdx.x * blockDim.x + threadIdx.x;     // NG*NC = 640
    if (i >= NG * NC) return;
    int g = i / NC, cc = i % NC;
    float inv = 1.0f / fmaxf(counts[g], 1.0f);
    float acc = lb[cc];
    for (int d = 0; d < DM; ++d)
        acc += sums[(size_t)g * DM + d] * inv * lw[(size_t)d * NC + cc];
    out[i] = acc;
}

// ------------------------------------------------------------------ launch
extern "C" void kernel_launch(void* const* d_in, const int* in_sizes, int n_in,
                              void* d_out, int out_size, void* d_ws, size_t ws_size,
                              hipStream_t stream) {
    const float* x0    = (const float*)d_in[0];
    const int*   ei    = (const int*)d_in[1];
    const int*   batch = (const int*)d_in[2];
    const float* Wrel  = (const float*)d_in[3];
    const float* brel  = (const float*)d_in[4];
    const float* Wroot = (const float*)d_in[5];
    const float* gamma = (const float*)d_in[6];
    const float* beta  = (const float*)d_in[7];
    const float* lw    = (const float*)d_in[8];
    const float* lb    = (const float*)d_in[9];
    const int* src = ei;
    const int* dst = ei + NE;
    float* out = (float*)d_out;

    // workspace carve-up (256B aligned slabs)
    size_t off = 0;
    char* base = (char*)d_ws;
    auto carve = [&](size_t bytes) -> char* {
        char* p = base + off;
        off += (bytes + 255) & ~(size_t)255;
        return p;
    };
    float*  agg   = (float*)carve((size_t)NND * DM * 4);
    float*  hbuf  = (float*)carve((size_t)NND * DM * 4);
    float*  xbuf  = (float*)carve((size_t)NND * DM * 4);
    bf16_t* pack  = (bf16_t*)carve((size_t)12 * (1 << 18) * sizeof(bf16_t));
    float*  stats = (float*)carve((size_t)2 * DM * 4);   // colsum | colsq
    float*  scale = (float*)carve((size_t)DM * 4);
    float*  shift = (float*)carve((size_t)DM * 4);
    float*  psums = (float*)carve((size_t)(NG * DM + NG + 4) * 4); // sums | counts
    float*  colsum = stats, *colsq = stats + DM;
    float*  counts = psums + NG * DM;

    // pack weights to bf16 fragment layout (cheap, deterministic each call)
    pack_all<<<12 * (1 << 18) / 256, 256, 0, stream>>>(Wrel, Wroot, pack);

    // zero pooling buffers
    zero_f4<<<((NG * DM + NG + 4) / 4 + 255) / 256, 256, 0, stream>>>(
        (float4*)psums, (NG * DM + NG + 4) / 4);

    const float* xcur = x0;
    for (int l = 0; l < NL; ++l) {
        zero_f4<<<(NND * DM / 4 + 255) / 256, 256, 0, stream>>>(
            (float4*)agg, NND * DM / 4);
        zero_f4<<<1, 256, 0, stream>>>((float4*)stats, 2 * DM / 4);

        edge_scatter<<<NE * 128 / 256, 256, 0, stream>>>(xcur, src, dst, agg);

        gemm_wmma<<<(NND / 16) * 4, 256, 0, stream>>>(
            agg, xcur,
            pack + (size_t)l * (1 << 18),           // W_rel[l]
            pack + (size_t)(6 + l) * (1 << 18),     // W_root[l]
            brel + (size_t)l * DM, hbuf, colsum, colsq);

        bn_finalize<<<(DM + 255) / 256, 256, 0, stream>>>(
            colsum, colsq, gamma + (size_t)l * DM, beta + (size_t)l * DM,
            scale, shift);

        bn_apply<<<(NND * DM / 4 + 255) / 256, 256, 0, stream>>>(
            hbuf, scale, shift, xbuf);
        xcur = xbuf;
    }

    pool_accum<<<NND * 128 / 256, 256, 0, stream>>>(xcur, batch, psums, counts);
    head<<<(NG * NC + 255) / 256, 256, 0, stream>>>(psums, counts, lw, lb, out);
}
